// GAT_18717467476274
// MI455X (gfx1250) — compile-verified, matrix operand order checked by reading, not run on hardware
//
#include <hip/hip_runtime.h>
#include <stdint.h>

typedef _Float16 v16h __attribute__((ext_vector_type(16)));
typedef _Float16 v8h  __attribute__((ext_vector_type(8)));
typedef float    v8f  __attribute__((ext_vector_type(8)));

#define LRELU_SLOPE 0.2f

__device__ __forceinline__ int f2ord(float f) {
  int i = __float_as_int(f);
  return (i >= 0) ? i : (i ^ 0x7FFFFFFF);
}
__device__ __forceinline__ float ord2f(int i) {
  return __int_as_float((i >= 0) ? i : (i ^ 0x7FFFFFFF));
}

// ---------------- init ----------------
__global__ void k_init_f32(float* p, float v, int n) {
  int i = blockIdx.x * blockDim.x + threadIdx.x;
  if (i < n) p[i] = v;
}
__global__ void k_init_i32(int* p, int v, int n) {
  int i = blockIdx.x * blockDim.x + threadIdx.x;
  if (i < n) p[i] = v;
}

// ---------------- edge list (both directions + self loops) ----------------
__global__ void k_build_edges(const long long* __restrict__ eg, int* __restrict__ src,
                              int* __restrict__ dst, int E0, int N) {
  int i = blockIdx.x * blockDim.x + threadIdx.x;
  int E = 2 * E0 + N;
  if (i >= E) return;
  int s, d;
  if (i < E0)            { s = (int)eg[i];          d = (int)eg[E0 + i]; }
  else if (i < 2 * E0)   { int j = i - E0; s = (int)eg[E0 + j]; d = (int)eg[j]; }
  else                   { s = d = i - 2 * E0; }
  src[i] = s; dst[i] = d;
}

// ---------------- dtype prep ----------------
__global__ void k_f32_to_f16(const float* __restrict__ in, _Float16* __restrict__ out, int n) {
  int i = blockIdx.x * blockDim.x + threadIdx.x;
  if (i < n) out[i] = (_Float16)in[i];
}

// Bt[(h*F+f)*Din + k] = W[h*Din*F + k*F + f]  (f16)
__global__ void k_prep_bt(const float* __restrict__ W, _Float16* __restrict__ Bt,
                          int Din, int F, int H) {
  int i = blockIdx.x * blockDim.x + threadIdx.x;
  int total = H * F * Din;
  if (i >= total) return;
  int k = i % Din;
  int c = i / Din;
  int h = c / F, f = c % F;
  Bt[i] = (_Float16)W[(size_t)h * Din * F + (size_t)k * F + f];
}

// ---------------- WMMA GEMM: C[MxNc] = A[MxK] * Bt[NcxK]^T, fp32 accum ----------------
// One wave per 16x16 tile; K multiple of 32; M,Nc multiples of 16.
__global__ void k_wmma_gemm(const _Float16* __restrict__ A, const _Float16* __restrict__ Bt,
                            float* __restrict__ C, int M, int Nc, int K) {
  int gt = blockIdx.x * blockDim.x + threadIdx.x;
  int wave = gt >> 5;
  int lane = threadIdx.x & 31;
  int tilesN = Nc >> 4;
  int tiles = (M >> 4) * tilesN;
  if (wave >= tiles) return;
  int tm = wave / tilesN, tn = wave % tilesN;
  int hid = lane >> 4;       // half of the wave: selects K sub-block
  int l16 = lane & 15;
  int m = tm * 16 + l16;     // A row for this lane
  int n = tn * 16 + l16;     // B column for this lane
  const _Float16* Arow = A + (size_t)m * K;
  const _Float16* Brow = Bt + (size_t)n * K;
  v8f c = {};
  for (int kc = 0; kc < K; kc += 32) {
    // 16-bit A/B VGPR layout: lanes 0-15 hold K = kc+0..7 and kc+16..23,
    // lanes 16-31 hold K = kc+8..15 and kc+24..31.
    v8h a0 = *(const v8h*)(Arow + kc + hid * 8);
    v8h a1 = *(const v8h*)(Arow + kc + 16 + hid * 8);
    v8h b0 = *(const v8h*)(Brow + kc + hid * 8);
    v8h b1 = *(const v8h*)(Brow + kc + 16 + hid * 8);
    v16h a, b;
#pragma unroll
    for (int e = 0; e < 8; ++e) {
      a[e] = a0[e]; a[e + 8] = a1[e];
      b[e] = b0[e]; b[e + 8] = b1[e];
    }
    c = __builtin_amdgcn_wmma_f32_16x16x32_f16(false, a, false, b, (short)0, c, false, false);
  }
  // D layout: VGPR v -> row tm*16 + v + 8*hid, col tn*16 + l16
#pragma unroll
  for (int v = 0; v < 8; ++v) {
    C[(size_t)(tm * 16 + v + 8 * hid) * Nc + tn * 16 + l16] = c[v];
  }
}

// ---------------- attention coefficients ----------------
// as[n*H+h] = dot(proj[n,h,:], a[h,0:F]);  an[n*H+h] = dot(proj[n,h,:], a[h,F:2F])
__global__ void k_alpha(const float* __restrict__ proj, const float* __restrict__ a,
                        float* __restrict__ as_, float* __restrict__ an_, int N, int H, int F) {
  int t = blockIdx.x * blockDim.x + threadIdx.x;
  if (t >= N * H) return;
  int n = t / H, h = t % H;
  const float* p  = proj + (size_t)n * H * F + (size_t)h * F;
  const float* av = a + (size_t)h * 2 * F;
  float s = 0.f, q = 0.f;
  for (int f = 0; f < F; ++f) {
    float pv = p[f];
    s += pv * av[f];
    q += pv * av[F + f];
  }
  as_[t] = s; an_[t] = q;
}

// ---------------- softmax passes over edges ----------------
__global__ void k_edge_logit(const int* __restrict__ src, const int* __restrict__ dst,
                             const float* __restrict__ as_, const float* __restrict__ an_,
                             float* __restrict__ eraw, int* __restrict__ emax, int E, int H) {
  int t = blockIdx.x * blockDim.x + threadIdx.x;
  if (t >= E * H) return;
  int i = t / H, h = t % H;
  int s = src[i], d = dst[i];
  float e = as_[d * H + h] + an_[s * H + h];
  e = (e > 0.f) ? e : LRELU_SLOPE * e;
  eraw[t] = e;
  atomicMax(&emax[d * H + h], f2ord(e));
}

__global__ void k_edge_exp(const int* __restrict__ dst, float* __restrict__ eexp,
                           const int* __restrict__ emax, float* __restrict__ den, int E, int H) {
  int t = blockIdx.x * blockDim.x + threadIdx.x;
  if (t >= E * H) return;
  int i = t / H, h = t % H;
  int d = dst[i];
  float ex = __expf(eexp[t] - ord2f(emax[d * H + h]));
  eexp[t] = ex;
  atomicAdd(&den[d * H + h], ex);
}

// ---------------- layer-1 aggregation: wave per edge, H=8, F=64, D=512 ----------------
__global__ void k_edge_agg_l1(const int* __restrict__ src, const int* __restrict__ dst,
                              const float* __restrict__ proj, const float* __restrict__ eexp,
                              const float* __restrict__ den, float* __restrict__ agg, int E) {
  int gt = blockIdx.x * blockDim.x + threadIdx.x;
  int wave = gt >> 5, lane = gt & 31;
  if (wave >= E) return;
  int s = src[wave], d = dst[wave];
  float wv = 0.f;
  if (lane < 8) wv = eexp[(size_t)wave * 8 + lane] / den[(size_t)d * 8 + lane];
  const float* ps = proj + (size_t)s * 512;
  float* ad = agg + (size_t)d * 512;
#pragma unroll
  for (int it = 0; it < 16; ++it) {
    float w = __shfl(wv, it >> 1, 32);  // head h = (it*32+lane)/64 = it/2, uniform per it
    int j = it * 32 + lane;
    atomicAdd(&ad[j], ps[j] * w);
  }
}

// ---------------- generic aggregation (layer 2, H=1): thread per (edge,f) ----------------
__global__ void k_edge_agg_fine(const int* __restrict__ src, const int* __restrict__ dst,
                                const float* __restrict__ proj, const float* __restrict__ eexp,
                                const float* __restrict__ den, float* __restrict__ out,
                                int E, int F) {
  int t = blockIdx.x * blockDim.x + threadIdx.x;
  if (t >= E * F) return;
  int i = t / F, f = t % F;
  int s = src[i], d = dst[i];
  float w = eexp[i] / den[d];
  atomicAdd(&out[(size_t)d * F + f], proj[(size_t)s * F + f] * w);
}

// ---------------- ELU + f16 convert between layers ----------------
__global__ void k_elu_f16(const float* __restrict__ in, _Float16* __restrict__ out, int n) {
  int i = blockIdx.x * blockDim.x + threadIdx.x;
  if (i >= n) return;
  float v = in[i];
  out[i] = (_Float16)((v > 0.f) ? v : (__expf(v) - 1.f));
}

extern "C" void kernel_launch(void* const* d_in, const int* in_sizes, int n_in,
                              void* d_out, int out_size, void* d_ws, size_t ws_size,
                              hipStream_t stream) {
  (void)n_in; (void)out_size; (void)ws_size;
  const float*     x     = (const float*)d_in[0];
  const long long* edges = (const long long*)d_in[1];
  const float*     W1    = (const float*)d_in[2];
  const float*     a1    = (const float*)d_in[3];
  const float*     W2    = (const float*)d_in[4];
  const float*     a2    = (const float*)d_in[5];

  const int Din = 128, F1 = 64, H1 = 8, Dmid = H1 * F1 /*512*/, F2 = 64;
  const int N  = in_sizes[0] / Din;   // 10000 (multiple of 16)
  const int E0 = in_sizes[1] / 2;     // 80000
  const int E  = 2 * E0 + N;          // 170000

  // ---- workspace bump allocator (256B aligned) ----
  char* wp = (char*)d_ws;
  auto alloc = [&](size_t bytes) -> char* {
    char* p = wp;
    wp += (bytes + 255) & ~(size_t)255;
    return p;
  };
  int*      srcv  = (int*)alloc((size_t)E * 4);
  int*      dstv  = (int*)alloc((size_t)E * 4);
  _Float16* xh    = (_Float16*)alloc((size_t)N * Din * 2);
  _Float16* bt1   = (_Float16*)alloc((size_t)Dmid * Din * 2);
  _Float16* bt2   = (_Float16*)alloc((size_t)F2 * Dmid * 2);
  float*    proj1 = (float*)alloc((size_t)N * Dmid * 4);
  float*    as1   = (float*)alloc((size_t)N * H1 * 4);
  float*    an1   = (float*)alloc((size_t)N * H1 * 4);
  int*      emax1 = (int*)alloc((size_t)N * H1 * 4);
  float*    den1  = (float*)alloc((size_t)N * H1 * 4);
  float*    eexp1 = (float*)alloc((size_t)E * H1 * 4);
  float*    agg1  = (float*)alloc((size_t)N * Dmid * 4);
  _Float16* hh    = (_Float16*)alloc((size_t)N * Dmid * 2);
  float*    proj2 = (float*)alloc((size_t)N * F2 * 4);
  float*    as2   = (float*)alloc((size_t)N * 4);
  float*    an2   = (float*)alloc((size_t)N * 4);
  int*      emax2 = (int*)alloc((size_t)N * 4);
  float*    den2  = (float*)alloc((size_t)N * 4);
  float*    eexp2 = (float*)alloc((size_t)E * 4);
  float*    out   = (float*)d_out;

  const int B = 256;
  auto cdiv = [](long long a, long long b) { return (int)((a + b - 1) / b); };
  const int ORD_NEG_INF = 0x807FFFFF;  // f2ord(-inf)

  // ---- prep ----
  k_build_edges<<<cdiv(E, B), B, 0, stream>>>(edges, srcv, dstv, E0, N);
  k_f32_to_f16<<<cdiv((long long)N * Din, B), B, 0, stream>>>(x, xh, N * Din);
  k_prep_bt<<<cdiv((long long)Dmid * Din, B), B, 0, stream>>>(W1, bt1, Din, F1, H1);
  k_prep_bt<<<cdiv((long long)F2 * Dmid, B), B, 0, stream>>>(W2, bt2, Dmid, F2, 1);

  // ---- layer 1 ----
  {
    int tiles = (N / 16) * (Dmid / 16);
    k_wmma_gemm<<<cdiv(tiles, 4), 128, 0, stream>>>(xh, bt1, proj1, N, Dmid, Din);
  }
  k_alpha<<<cdiv((long long)N * H1, B), B, 0, stream>>>(proj1, a1, as1, an1, N, H1, F1);
  k_init_i32<<<cdiv(N * H1, B), B, 0, stream>>>(emax1, ORD_NEG_INF, N * H1);
  k_init_f32<<<cdiv(N * H1, B), B, 0, stream>>>(den1, 0.f, N * H1);
  k_init_f32<<<cdiv((long long)N * Dmid, B), B, 0, stream>>>(agg1, 0.f, N * Dmid);
  k_edge_logit<<<cdiv((long long)E * H1, B), B, 0, stream>>>(srcv, dstv, as1, an1, eexp1, emax1, E, H1);
  k_edge_exp<<<cdiv((long long)E * H1, B), B, 0, stream>>>(dstv, eexp1, emax1, den1, E, H1);
  k_edge_agg_l1<<<cdiv((long long)E * 32, B), B, 0, stream>>>(srcv, dstv, proj1, eexp1, den1, agg1, E);
  k_elu_f16<<<cdiv((long long)N * Dmid, B), B, 0, stream>>>(agg1, hh, N * Dmid);

  // ---- layer 2 ----
  {
    int tiles = (N / 16) * (F2 / 16);
    k_wmma_gemm<<<cdiv(tiles, 4), 128, 0, stream>>>(hh, bt2, proj2, N, F2, Dmid);
  }
  k_alpha<<<cdiv(N, B), B, 0, stream>>>(proj2, a2, as2, an2, N, 1, F2);
  k_init_i32<<<cdiv(N, B), B, 0, stream>>>(emax2, ORD_NEG_INF, N);
  k_init_f32<<<cdiv(N, B), B, 0, stream>>>(den2, 0.f, N);
  k_init_f32<<<cdiv((long long)N * F2, B), B, 0, stream>>>(out, 0.f, N * F2);
  k_edge_logit<<<cdiv(E, B), B, 0, stream>>>(srcv, dstv, as2, an2, eexp2, emax2, E, 1);
  k_edge_exp<<<cdiv(E, B), B, 0, stream>>>(dstv, eexp2, emax2, den2, E, 1);
  k_edge_agg_fine<<<cdiv((long long)E * F2, B), B, 0, stream>>>(srcv, dstv, proj2, eexp2, den2, out, E, F2);
}
